// MoE_7851200217347
// MI455X (gfx1250) — matrix operand order repaired and checked by reading, not compiled, not run on hardware
//
#include <hip/hip_runtime.h>
#include <hip/hip_bf16.h>
#include <math.h>

// Problem constants (from reference setup_inputs)
#define DD 1024   // model dim
#define FF 4096   // ffn dim
#define EE 8      // experts
#define NN 8192   // tokens (B*T = 4*2048)
#define TM 32     // token tile (2 x WMMA M)
#define FCC 128   // F chunk per pipeline stage
#define XP (DD + 8)   // padded LDS row stride (bank-conflict-free A loads)
#define HP (FCC + 8)  // padded LDS row stride for H chunk

typedef __bf16 bf16_t;
typedef bf16_t bf16x16 __attribute__((ext_vector_type(16)));
typedef float  f32x8   __attribute__((ext_vector_type(8)));

__device__ __forceinline__ unsigned short f2bf(float f) {
  union { float f; unsigned int u; } v; v.f = f;
  unsigned int r = v.u + 0x7FFFu + ((v.u >> 16) & 1u);  // round-to-nearest-even
  return (unsigned short)(r >> 16);
}

__device__ __forceinline__ bf16x16 make_frag(uint4 lo, uint4 hi) {
  union { uint4 u[2]; bf16x16 v; } x;
  x.u[0] = lo; x.u[1] = hi;
  return x.v;
}

// ---------------------------------------------------------------- cvt x -> bf16
__global__ void cvt_x_kernel(const float* __restrict__ x,
                             unsigned short* __restrict__ xb, long n8) {
  long i = (long)blockIdx.x * blockDim.x + threadIdx.x;  // 8 elems / thread
  if (i >= n8) return;
  const float4* src = (const float4*)x;
  float4 a = src[i * 2], b = src[i * 2 + 1];
  uint4 o;
  o.x = (unsigned)f2bf(a.x) | ((unsigned)f2bf(a.y) << 16);
  o.y = (unsigned)f2bf(a.z) | ((unsigned)f2bf(a.w) << 16);
  o.z = (unsigned)f2bf(b.x) | ((unsigned)f2bf(b.y) << 16);
  o.w = (unsigned)f2bf(b.z) | ((unsigned)f2bf(b.w) << 16);
  ((uint4*)xb)[i] = o;
}

// --------------------------------------- per-expert transpose fp32[R][C] -> bf16[C][R]
__global__ void transpose_cvt_kernel(const float* __restrict__ src,
                                     unsigned short* __restrict__ dst,
                                     int R, int C) {
  __shared__ float tile[32][33];
  long per = (long)R * C;
  src += (long)blockIdx.z * per;
  dst += (long)blockIdx.z * per;
  int c0 = blockIdx.x * 32, r0 = blockIdx.y * 32;
  int tx = threadIdx.x & 31, ty = threadIdx.x >> 5;
#pragma unroll
  for (int j = 0; j < 4; ++j)
    tile[ty + 8 * j][tx] = src[(long)(r0 + ty + 8 * j) * C + c0 + tx];
  __syncthreads();
#pragma unroll
  for (int j = 0; j < 4; ++j)
    dst[(long)(c0 + ty + 8 * j) * R + r0 + tx] = f2bf(tile[tx][ty + 8 * j]);
}

// ---------------------------------------------------------------- gating / top-2
__global__ void gate_kernel(const float* __restrict__ x,
                            const float* __restrict__ Wg,
                            const float* __restrict__ bg,
                            int* __restrict__ pair_tok,
                            float* __restrict__ pair_w,
                            int* __restrict__ counts) {
  int lane = threadIdx.x & 31;
  int n = blockIdx.x * 8 + (threadIdx.x >> 5);  // one wave32 per token
  float s[8];
#pragma unroll
  for (int e = 0; e < 8; ++e) s[e] = 0.f;
  const float* xr = x + (long)n * DD;
  for (int d = lane; d < DD; d += 32) {
    float xv = xr[d];
    float4 w0 = *(const float4*)(Wg + (long)d * 8);
    float4 w1 = *(const float4*)(Wg + (long)d * 8 + 4);
    s[0] += xv * w0.x; s[1] += xv * w0.y; s[2] += xv * w0.z; s[3] += xv * w0.w;
    s[4] += xv * w1.x; s[5] += xv * w1.y; s[6] += xv * w1.z; s[7] += xv * w1.w;
  }
#pragma unroll
  for (int e = 0; e < 8; ++e) {
#pragma unroll
    for (int off = 16; off > 0; off >>= 1) s[e] += __shfl_xor(s[e], off, 32);
  }
  if (lane == 0) {
#pragma unroll
    for (int e = 0; e < 8; ++e) s[e] += bg[e];
    int i0 = 0;
#pragma unroll
    for (int e = 1; e < 8; ++e) if (s[e] > s[i0]) i0 = e;
    int i1 = (i0 == 0) ? 1 : 0;
#pragma unroll
    for (int e = 0; e < 8; ++e) if (e != i0 && s[e] > s[i1]) i1 = e;
    float e1  = expf(s[i1] - s[i0]);   // softmax over {top1, top2}
    float inv = 1.f / (1.f + e1);
    int s0 = atomicAdd(&counts[i0], 1);
    pair_tok[i0 * NN + s0] = n; pair_w[i0 * NN + s0] = inv;
    int s1 = atomicAdd(&counts[i1], 1);
    pair_tok[i1 * NN + s1] = n; pair_w[i1 * NN + s1] = e1 * inv;
  }
}

// ---------------------------------------------------------------- fused expert FFN
// Block = (expert e, 32 tokens of e's list). 8 waves, each owns a 32x128 out strip
// (16 WMMA accumulator tiles = 128 VGPRs). B fragments are shared by 2 M-tiles and
// software-prefetched one step ahead so an L2 load stays in flight under the WMMAs.
__global__ void __launch_bounds__(256, 1)
moe_ffn_kernel(const unsigned short* __restrict__ Xb,    // [N][D] bf16
               const unsigned short* __restrict__ W1t,   // [E][F][D] bf16
               const unsigned short* __restrict__ W2t,   // [E][D][F] bf16
               const float* __restrict__ b1,             // [E][F]
               const float* __restrict__ b2,             // [E][D]
               const int* __restrict__ pair_tok,
               const float* __restrict__ pair_w,
               const int* __restrict__ counts,
               float* __restrict__ out) {
  __shared__ __align__(16) unsigned short Xs[TM * XP];   // ~66 KB, gathered X tile
  __shared__ __align__(16) unsigned short Hs[TM * HP];   // ~8.7 KB, gelu(H) chunk
  __shared__ int   tok_s[TM];
  __shared__ float wgt_s[TM];

  const int e   = blockIdx.y;
  const int cnt = counts[e];
  const int m0  = blockIdx.x * TM;
  if (m0 >= cnt) return;  // uniform early-exit

  const int tid  = threadIdx.x;
  const int w    = tid >> 5;
  const int lane = tid & 31;
  const int col  = lane & 15;              // N index (and A-row index)
  const int hi   = (lane >= 16);
  const int kbA  = hi ? 8 : 0;             // A frag: K 0-7/16-23 vs 8-15/24-31
  const int kbB  = hi ? 8 : 0;             // B frag: K 0-7/16-23 vs 8-15/24-31
  const int mb   = hi ? 8 : 0;             // C/D: M = r + 8*(lane>=16)

  if (tid < TM) {
    int i = m0 + tid;
    if (i < cnt) { tok_s[tid] = pair_tok[e * NN + i]; wgt_s[tid] = pair_w[e * NN + i]; }
    else         { tok_s[tid] = 0;                    wgt_s[tid] = 0.f; }
  }
  __syncthreads();

  { // gather 32 token rows (bf16) into LDS, coalesced b128
    int r = tid >> 3, c = tid & 7;
    const uint4* src = (const uint4*)(Xb + (long)tok_s[r] * DD);
    uint4* dst = (uint4*)(Xs + r * XP);
#pragma unroll
    for (int j = 0; j < 16; ++j) dst[c + 8 * j] = src[c + 8 * j];
  }
  __syncthreads();

  f32x8 acc[16];                           // [mh*8 + n], mh in {0,1}
#pragma unroll
  for (int n = 0; n < 16; ++n)
#pragma unroll
    for (int r = 0; r < 8; ++r) acc[n][r] = 0.f;

  const unsigned short* xrow0 = Xs + col * XP + kbA;          // rows 0-15
  const unsigned short* xrow1 = Xs + (16 + col) * XP + kbA;   // rows 16-31
  const unsigned short* hrow0 = Hs + col * HP + kbA;
  const unsigned short* hrow1 = Hs + (16 + col) * HP + kbA;
  const unsigned short* w2p   = W2t + ((long)e * DD + w * 128 + col) * FF + kbB;

  for (int fc = 0; fc < FF; fc += FCC) {
    // ---- Phase 1: this wave's 32x16 slab of H = gelu(X @ W1 + b1)
    int f = fc + w * 16 + col;
    const unsigned short* w1p = W1t + ((long)e * FF + f) * DD + kbB;
    float bv = b1[e * FF + f];
    f32x8 h0, h1;
#pragma unroll
    for (int r = 0; r < 8; ++r) { h0[r] = bv; h1[r] = bv; }

    uint4 bl = *(const uint4*)(w1p);
    uint4 bh = *(const uint4*)(w1p + 16);
#pragma unroll 4
    for (int kc = 0; kc < DD - 32; kc += 32) {
      uint4 nbl = *(const uint4*)(w1p + kc + 32);      // prefetch next B
      uint4 nbh = *(const uint4*)(w1p + kc + 48);
      bf16x16 a0 = make_frag(*(const uint4*)&xrow0[kc], *(const uint4*)&xrow0[kc + 16]);
      bf16x16 a1 = make_frag(*(const uint4*)&xrow1[kc], *(const uint4*)&xrow1[kc + 16]);
      bf16x16 b  = make_frag(bl, bh);
      h0 = __builtin_amdgcn_wmma_f32_16x16x32_bf16(false, a0, false, b,
                                                   (short)0, h0, false, false);
      h1 = __builtin_amdgcn_wmma_f32_16x16x32_bf16(false, a1, false, b,
                                                   (short)0, h1, false, false);
      bl = nbl; bh = nbh;
    }
    { // last K step
      const int kc = DD - 32;
      bf16x16 a0 = make_frag(*(const uint4*)&xrow0[kc], *(const uint4*)&xrow0[kc + 16]);
      bf16x16 a1 = make_frag(*(const uint4*)&xrow1[kc], *(const uint4*)&xrow1[kc + 16]);
      bf16x16 b  = make_frag(bl, bh);
      h0 = __builtin_amdgcn_wmma_f32_16x16x32_bf16(false, a0, false, b,
                                                   (short)0, h0, false, false);
      h1 = __builtin_amdgcn_wmma_f32_16x16x32_bf16(false, a1, false, b,
                                                   (short)0, h1, false, false);
    }
#pragma unroll
    for (int r = 0; r < 8; ++r) {
      float v0 = h0[r];
      v0 = 0.5f * v0 * (1.f + erff(v0 * 0.70710678118654752f));  // exact GELU
      Hs[(mb + r) * HP + w * 16 + col] = f2bf(v0);
      float v1 = h1[r];
      v1 = 0.5f * v1 * (1.f + erff(v1 * 0.70710678118654752f));
      Hs[(16 + mb + r) * HP + w * 16 + col] = f2bf(v1);
    }
    __syncthreads();

    // ---- Phase 2: acc[32 x 128 strip] += H_chunk @ W2t (B prefetch rotation)
    uint4 cl = *(const uint4*)(w2p + fc);
    uint4 ch = *(const uint4*)(w2p + fc + 16);
    bf16x16 a0, a1;
#pragma unroll
    for (int idx = 0; idx < 32; ++idx) {
      const int kf = (idx >> 3) * 32;
      const int n  = idx & 7;
      if (n == 0) {
        a0 = make_frag(*(const uint4*)&hrow0[kf], *(const uint4*)&hrow0[kf + 16]);
        a1 = make_frag(*(const uint4*)&hrow1[kf], *(const uint4*)&hrow1[kf + 16]);
      }
      uint4 nl = cl, nh = ch;
      if (idx + 1 < 32) {                                  // compile-time branch
        const int kf2 = ((idx + 1) >> 3) * 32;
        const int n2  = (idx + 1) & 7;
        const unsigned short* q = w2p + fc + kf2 + (long)n2 * 16 * FF;
        nl = *(const uint4*)(q);
        nh = *(const uint4*)(q + 16);
      }
      bf16x16 b = make_frag(cl, ch);
      acc[n]     = __builtin_amdgcn_wmma_f32_16x16x32_bf16(false, a0, false, b,
                                                           (short)0, acc[n], false, false);
      acc[8 + n] = __builtin_amdgcn_wmma_f32_16x16x32_bf16(false, a1, false, b,
                                                           (short)0, acc[8 + n], false, false);
      cl = nl; ch = nh;
    }
    __syncthreads();
  }

  // ---- Epilogue: +b2, scale by gate weight, atomic combine into out
#pragma unroll
  for (int mh = 0; mh < 2; ++mh) {
#pragma unroll
    for (int n = 0; n < 8; ++n) {
      int dcol = w * 128 + n * 16 + col;
      float b2v = b2[e * DD + dcol];
#pragma unroll
      for (int r = 0; r < 8; ++r) {
        int M = mh * 16 + mb + r;
        if (m0 + M < cnt) {
          float val = (acc[mh * 8 + n][r] + b2v) * wgt_s[M];
          unsafeAtomicAdd(&out[(long)tok_s[M] * DD + dcol], val);
        }
      }
    }
  }
}

// ---------------------------------------------------------------- launch
extern "C" void kernel_launch(void* const* d_in, const int* in_sizes, int n_in,
                              void* d_out, int out_size, void* d_ws, size_t ws_size,
                              hipStream_t stream) {
  const float* x  = (const float*)d_in[0];
  const float* Wg = (const float*)d_in[1];
  const float* bg = (const float*)d_in[2];
  const float* W1 = (const float*)d_in[3];
  const float* b1 = (const float*)d_in[4];
  const float* W2 = (const float*)d_in[5];
  const float* b2 = (const float*)d_in[6];
  // d_in[7] = top_k == 2 (hardcoded in the top-2 gate kernel)

  // Workspace layout (~145 MB): bf16 X + bf16 transposed weights (fits 192MB L2)
  char* ws = (char*)d_ws;
  size_t off = 0;
  unsigned short* Xb  = (unsigned short*)(ws + off); off += (size_t)NN * DD * 2;
  unsigned short* W1t = (unsigned short*)(ws + off); off += (size_t)EE * DD * FF * 2;
  unsigned short* W2t = (unsigned short*)(ws + off); off += (size_t)EE * FF * DD * 2;
  int*   ptok   = (int*)(ws + off);   off += (size_t)EE * NN * 4;
  float* pw     = (float*)(ws + off); off += (size_t)EE * NN * 4;
  int*   counts = (int*)(ws + off);   off += 256;
  (void)ws_size; (void)in_sizes; (void)n_in;

  float* outf = (float*)d_out;
  hipMemsetAsync(counts, 0, EE * sizeof(int), stream);
  hipMemsetAsync(outf, 0, (size_t)out_size * sizeof(float), stream);

  long n8 = (long)NN * DD / 8;
  cvt_x_kernel<<<(unsigned)((n8 + 255) / 256), 256, 0, stream>>>(x, Xb, n8);
  // W1 [E][D][F] -> W1t [E][F][D]
  transpose_cvt_kernel<<<dim3(FF / 32, DD / 32, EE), 256, 0, stream>>>(W1, W1t, DD, FF);
  // W2 [E][F][D] -> W2t [E][D][F]
  transpose_cvt_kernel<<<dim3(DD / 32, FF / 32, EE), 256, 0, stream>>>(W2, W2t, FF, DD);
  gate_kernel<<<NN / 8, 256, 0, stream>>>(x, Wg, bg, ptok, pw, counts);
  moe_ffn_kernel<<<dim3(NN / TM, EE), 256, 0, stream>>>(Xb, W1t, W2t, b1, b2,
                                                        ptok, pw, counts, outf);
}
